// MultiHeadAttentionWithRoPE_49529562858303
// MI455X (gfx1250) — compile-verified, matrix operand order checked by reading, not run on hardware
//
#include <hip/hip_runtime.h>
#include <hip/hip_bf16.h>
#include <stdint.h>

// ---------- types ----------
typedef __attribute__((ext_vector_type(4)))  __bf16 v4bf;
typedef __attribute__((ext_vector_type(8)))  __bf16 v8bf;
typedef __attribute__((ext_vector_type(16))) __bf16 v16bf;
typedef __attribute__((ext_vector_type(8)))  float  v8f;

struct BfPair { v8bf lo, hi; };

__device__ inline __bf16 f2bf(float f) {
    unsigned u = __builtin_bit_cast(unsigned, f);
    u += 0x7FFFu + ((u >> 16) & 1u);                 // round-to-nearest-even
    unsigned short h = (unsigned short)(u >> 16);
    return __builtin_bit_cast(__bf16, h);
}
__device__ inline float bf2f(__bf16 b) {
    unsigned short h = __builtin_bit_cast(unsigned short, b);
    return __builtin_bit_cast(float, ((unsigned)h) << 16);
}

// 16-bit fragment per CDNA5 ISA layout (A 16x32 / B 32x16 mirrored):
// lane = (row|col)%16, hi = lane/16; elems 0..7 <- K = hi*8.., elems 8..15 <- K = 16+hi*8..
__device__ inline v16bf load_frag(const __bf16* row, int hi) {
    BfPair p;
    p.lo = *(const v8bf*)(row + hi * 8);
    p.hi = *(const v8bf*)(row + 16 + hi * 8);
    return __builtin_bit_cast(v16bf, p);
}

__device__ inline v8f wmma_bf16(v16bf a, v16bf b, v8f c) {
    return __builtin_amdgcn_wmma_f32_16x16x32_bf16(
        false, a, false, b, (short)0, c, false, false);
}

// ---------- CDNA5 async global->LDS copy (ASYNCcnt-tracked) ----------
__device__ inline void async_b128(const __bf16* g, __bf16* l) {
    const unsigned lds = (unsigned)(uintptr_t)(l);              // low 32 bits = LDS offset
    const unsigned long long ga = (unsigned long long)(uintptr_t)(g);
    asm volatile("global_load_async_to_lds_b128 %0, %1, off"
                 :: "v"(lds), "v"(ga) : "memory");
}
__device__ inline void wait_async0() {
    asm volatile("s_wait_asynccnt 0x0" ::: "memory");
}

// ---------- fp32 -> bf16 pre-conversion ----------
__global__ __launch_bounds__(256) void cvt_f32_bf16(const float* __restrict__ in,
                                                    __bf16* __restrict__ out, int n)
{
    const int i = (blockIdx.x * 256 + threadIdx.x) * 4;
    if (i + 3 < n) {
        float4 v = *(const float4*)(in + i);
        v4bf o;
        o[0] = f2bf(v.x); o[1] = f2bf(v.y); o[2] = f2bf(v.z); o[3] = f2bf(v.w);
        *(v4bf*)(out + i) = o;
    }
}

// ---------- GEMM: Out = A[M,K](bf16) @ W[N,K](bf16)^T + bias ----------
// 128x128 tile / 256 threads (8 waves), K-step 64 (2 WMMA depths),
// double-buffered async LDS staging: 16 WMMAs per wave between barriers.
// OUT_MODE: 0 = f32 row-major, 1 = bf16 row-major, 2 = bf16 [b,h,d,s] (V-transposed)
constexpr int GP = 72; // LDS pitch (bf16) for 64-wide k-slice; chunks stay 16B-aligned

template <int OUT_MODE>
__global__ __launch_bounds__(256) void gemm_bf16_kernel(
    const __bf16* __restrict__ A, const __bf16* __restrict__ W,
    const float* __restrict__ bias, void* __restrict__ Out,
    int M, int N, int K)
{
    __shared__ __bf16 sA[2][128 * GP];
    __shared__ __bf16 sB[2][128 * GP];

    const int t      = threadIdx.x;
    const int lane   = t & 31;
    const int wave   = t >> 5;
    const int lane16 = lane & 15;
    const int hi     = lane >> 4;
    const int mOff   = (wave & 3) * 32;   // 2 m-subtiles per wave
    const int nOff   = (wave >> 2) * 64;  // 4 n-subtiles per wave
    const int mBase  = blockIdx.x * 128;
    const int nBase  = blockIdx.y * 128;

    const int srow = t >> 1;          // 0..127
    const int scol = (t & 1) * 32;    // 0 or 32
    const __bf16* gA = A + (size_t)(mBase + srow) * K + scol;
    const __bf16* gB = W + (size_t)(nBase + srow) * K + scol;

    auto issue = [&](int kt, int buf) {
        #pragma unroll
        for (int i = 0; i < 4; i++) {
            async_b128(gA + kt + i * 8, &sA[buf][srow * GP + scol + i * 8]);
            async_b128(gB + kt + i * 8, &sB[buf][srow * GP + scol + i * 8]);
        }
    };

    v8f acc[2][4];
    #pragma unroll
    for (int i = 0; i < 2; i++)
        #pragma unroll
        for (int j = 0; j < 4; j++) acc[i][j] = {};

    const int nk = K / 64;
    issue(0, 0);
    wait_async0();
    __syncthreads();

    for (int ki = 0; ki < nk; ++ki) {
        const int buf = ki & 1;
        if (ki + 1 < nk) issue((ki + 1) * 64, (ki + 1) & 1);  // prefetch next tile

        #pragma unroll
        for (int kk = 0; kk < 64; kk += 32) {
            v16bf af[2], bfr[4];
            #pragma unroll
            for (int i = 0; i < 2; i++)
                af[i] = load_frag(&sA[buf][(mOff + i * 16 + lane16) * GP + kk], hi);
            #pragma unroll
            for (int j = 0; j < 4; j++)
                bfr[j] = load_frag(&sB[buf][(nOff + j * 16 + lane16) * GP + kk], hi);
            #pragma unroll
            for (int i = 0; i < 2; i++)
                #pragma unroll
                for (int j = 0; j < 4; j++)
                    acc[i][j] = wmma_bf16(af[i], bfr[j], acc[i][j]);
        }

        wait_async0();       // prefetch landed
        __syncthreads();     // everyone done reading current buffer
    }

    // epilogue: + bias, store
    #pragma unroll
    for (int i = 0; i < 2; i++)
        #pragma unroll
        for (int j = 0; j < 4; j++) {
            const int col = nBase + nOff + j * 16 + lane16;
            const float bv = bias[col];
            #pragma unroll
            for (int v = 0; v < 8; v++) {
                const int row = mBase + mOff + i * 16 + v + 8 * hi; // C layout: M = v + 8*hi
                const float val = acc[i][j][v] + bv;
                if constexpr (OUT_MODE == 2) {
                    // V transposed per head: out[((b*16+h)*64+e)][s] = [b*1024+n][s]
                    const int b = row >> 11, s = row & 2047;
                    ((__bf16*)Out)[((size_t)(b * 1024 + col) << 11) + s] = f2bf(val);
                } else if constexpr (OUT_MODE == 1) {
                    ((__bf16*)Out)[(size_t)row * N + col] = f2bf(val);
                } else {
                    ((float*)Out)[(size_t)row * N + col] = val;
                }
            }
        }
}

// ---------- RoPE (in-place on bf16 [8192,1024]) ----------
// scaleOut folds the attention 1/sqrt(hd) into Q: softmax(QK^T/8)=softmax((Q/8)K^T)
__global__ __launch_bounds__(256) void rope_kernel(__bf16* __restrict__ X, float scaleOut)
{
    const int idx = blockIdx.x * 256 + threadIdx.x; // 8192*16*32 pairs
    const int e = idx & 31;
    const int h = (idx >> 5) & 15;
    const int m = idx >> 9;          // 0..8191
    const int tpos = m & 2047;       // seq position
    const size_t row = (size_t)m * 1024 + h * 64;

    const float x1 = bf2f(X[row + e]);
    const float x2 = bf2f(X[row + e + 32]);
    const float LN = 9.210340371976184f; // ln(10000)
    const int i = e >> 1;
    const float th1 = __expf(-(float)i        * (LN / 32.0f));
    const float th2 = __expf(-(float)(i + 16) * (LN / 32.0f));
    float s1, c1, s2, c2;
    __sincosf((float)tpos * th1, &s1, &c1);
    __sincosf((float)tpos * th2, &s2, &c2);
    X[row + e]      = f2bf((x1 * c1 - x2 * s1) * scaleOut);  // out[j]=x*cos+rot(x)*sin
    X[row + e + 32] = f2bf((x2 * c2 + x1 * s2) * scaleOut);
}

// ---------- Flash attention: bf16 WMMA, fp32 online softmax ----------
// block = 128 (4 waves), wave owns 16 full query rows. grid = (b*h, seq/64).
// K from [8192,1024] row-major; V from [b,h,d,s] transposed layout.
// KV tiles double-buffered via async LDS copies. Q is pre-scaled by 1/8.
__global__ __launch_bounds__(128) void flash_attn_kernel(
    const __bf16* __restrict__ Q, const __bf16* __restrict__ K,
    const __bf16* __restrict__ Vt, __bf16* __restrict__ O)
{
    constexpr int S = 2048, D = 1024, HD = 64, P = 72;
    __shared__ __bf16 sQ[64 * P];
    __shared__ __bf16 sK[2][64 * P];
    __shared__ __bf16 sVt[2][64 * P];   // [d][kv]
    __shared__ __bf16 sP[4 * 16 * P];   // wave-private P strips

    const int bh = blockIdx.x;
    const int b = bh >> 4, h = bh & 15;
    const int qBase = blockIdx.y * 64;
    const int t = threadIdx.x, lane = t & 31, w = t >> 5;
    const int lane16 = lane & 15, hi = lane >> 4;
    const size_t headBase = (size_t)b * S * D + (size_t)h * HD;
    const size_t vtBase = (size_t)bh * HD * S;   // [bh][64][2048]

    const int r  = t >> 1;          // staging row 0..63
    const int c0 = (t & 1) * 32;    // staging col 0 or 32

    // stage Q tile 64x64 (async)
    {
        const __bf16* gp = Q + headBase + (size_t)(qBase + r) * D + c0;
        #pragma unroll
        for (int i = 0; i < 4; i++)
            async_b128(gp + i * 8, &sQ[r * P + c0 + i * 8]);
    }

    auto issueKV = [&](int kt, int buf) {
        const __bf16* kp = K + headBase + (size_t)(kt + r) * D + c0;
        const __bf16* vp = Vt + vtBase + (size_t)r * S + kt + c0;
        #pragma unroll
        for (int i = 0; i < 4; i++) {
            async_b128(kp + i * 8, &sK[buf][r * P + c0 + i * 8]);
            async_b128(vp + i * 8, &sVt[buf][r * P + c0 + i * 8]);
        }
    };

    issueKV(0, 0);
    wait_async0();
    __syncthreads();

    const v16bf qf0 = load_frag(&sQ[(w * 16 + lane16) * P], hi);
    const v16bf qf1 = load_frag(&sQ[(w * 16 + lane16) * P + 32], hi);
    __bf16* sPw = &sP[w * 16 * P];

    v8f of[4];
    #pragma unroll
    for (int j = 0; j < 4; j++) of[j] = {};
    float mH[8], L[8];
    #pragma unroll
    for (int v = 0; v < 8; v++) { mH[v] = -1e30f; L[v] = 0.0f; }

    for (int ki = 0; ki < S / 64; ++ki) {
        const int buf = ki & 1;
        if (ki + 1 < S / 64) issueKV((ki + 1) * 64, (ki + 1) & 1);

        // S = (Q/8) K^T  -- scale pre-folded into Q
        v8f sacc[4];
        #pragma unroll
        for (int j = 0; j < 4; j++) {
            v16bf kb0 = load_frag(&sK[buf][(j * 16 + lane16) * P], hi);
            v16bf kb1 = load_frag(&sK[buf][(j * 16 + lane16) * P + 32], hi);
            v8f z = {};
            z = wmma_bf16(qf0, kb0, z);
            z = wmma_bf16(qf1, kb1, z);
            sacc[j] = z;
        }

        // online softmax (rows m = v + 8*hi; 16-lane shfl reductions)
        #pragma unroll
        for (int v = 0; v < 8; v++) {
            float rm = -1e30f;
            #pragma unroll
            for (int j = 0; j < 4; j++) rm = fmaxf(rm, sacc[j][v]);
            rm = fmaxf(rm, __shfl_xor(rm, 1));
            rm = fmaxf(rm, __shfl_xor(rm, 2));
            rm = fmaxf(rm, __shfl_xor(rm, 4));
            rm = fmaxf(rm, __shfl_xor(rm, 8));
            const float newM = fmaxf(mH[v], rm);
            const float f = __expf(mH[v] - newM);
            float ps = 0.0f;
            #pragma unroll
            for (int j = 0; j < 4; j++) {
                const float p = __expf(sacc[j][v] - newM);
                ps += p;
                sPw[(v + 8 * hi) * P + j * 16 + lane16] = f2bf(p);
            }
            ps += __shfl_xor(ps, 1);
            ps += __shfl_xor(ps, 2);
            ps += __shfl_xor(ps, 4);
            ps += __shfl_xor(ps, 8);
            L[v] = L[v] * f + ps;
            mH[v] = newM;
            #pragma unroll
            for (int j = 0; j < 4; j++) of[j][v] *= f;
        }

        // O += P V (sPw wave-private; in-wave LDS ordering suffices)
        const v16bf pa0 = load_frag(&sPw[lane16 * P], hi);
        const v16bf pa1 = load_frag(&sPw[lane16 * P + 32], hi);
        #pragma unroll
        for (int j = 0; j < 4; j++) {
            v16bf vb0 = load_frag(&sVt[buf][(j * 16 + lane16) * P], hi);
            v16bf vb1 = load_frag(&sVt[buf][(j * 16 + lane16) * P + 32], hi);
            of[j] = wmma_bf16(pa0, vb0, of[j]);
            of[j] = wmma_bf16(pa1, vb1, of[j]);
        }

        wait_async0();     // next KV tile landed
        __syncthreads();   // all waves done with current buffers
    }

    // normalize + store O as bf16 [8192,1024]
    #pragma unroll
    for (int j = 0; j < 4; j++)
        #pragma unroll
        for (int v = 0; v < 8; v++) {
            const int qr = qBase + w * 16 + v + 8 * hi;
            const int dc = j * 16 + lane16;
            O[headBase + (size_t)qr * D + dc] = f2bf(of[j][v] / L[v]);
        }
}

// ---------- host launch ----------
extern "C" void kernel_launch(void* const* d_in, const int* in_sizes, int n_in,
                              void* d_out, int out_size, void* d_ws, size_t ws_size,
                              hipStream_t stream) {
    (void)in_sizes; (void)n_in; (void)out_size; (void)ws_size;
    const float* x  = (const float*)d_in[0];
    const float* wq = (const float*)d_in[1];
    const float* bq = (const float*)d_in[2];
    const float* wk = (const float*)d_in[3];
    const float* bk = (const float*)d_in[4];
    const float* wv = (const float*)d_in[5];
    const float* bv = (const float*)d_in[6];
    const float* wo = (const float*)d_in[7];
    const float* bo = (const float*)d_in[8];
    float* out = (float*)d_out;

    const int M = 8192, N = 1024, K = 1024;
    const size_t MD = (size_t)M * N;     // 8M elems
    const size_t WD = (size_t)N * K;     // 1M elems

    __bf16* xbf  = (__bf16*)d_ws;        // [8192,1024]
    __bf16* wqb  = xbf + MD;             // [1024,1024] each
    __bf16* wkb  = wqb + WD;
    __bf16* wvb  = wkb + WD;
    __bf16* wob  = wvb + WD;
    __bf16* qws  = wob + WD;             // [8192,1024]
    __bf16* kws  = qws + MD;             // [8192,1024]
    __bf16* vtws = kws + MD;             // [64 bh][64 d][2048 s]
    __bf16* ows  = vtws + MD;            // [8192,1024]  (total 88 MB)

    // pre-convert activations + weights to bf16
    cvt_f32_bf16<<<(int)(MD / 1024), 256, 0, stream>>>(x,  xbf, (int)MD);
    cvt_f32_bf16<<<(int)(WD / 1024), 256, 0, stream>>>(wq, wqb, (int)WD);
    cvt_f32_bf16<<<(int)(WD / 1024), 256, 0, stream>>>(wk, wkb, (int)WD);
    cvt_f32_bf16<<<(int)(WD / 1024), 256, 0, stream>>>(wv, wvb, (int)WD);
    cvt_f32_bf16<<<(int)(WD / 1024), 256, 0, stream>>>(wo, wob, (int)WD);

    dim3 gGemm(M / 128, N / 128);
    gemm_bf16_kernel<1><<<gGemm, 256, 0, stream>>>(xbf, wqb, bq, qws, M, N, K);
    gemm_bf16_kernel<1><<<gGemm, 256, 0, stream>>>(xbf, wkb, bk, kws, M, N, K);
    gemm_bf16_kernel<2><<<gGemm, 256, 0, stream>>>(xbf, wvb, bv, vtws, M, N, K);

    rope_kernel<<<(8192 * 16 * 32) / 256, 256, 0, stream>>>(qws, 0.125f); // fold 1/sqrt(64)
    rope_kernel<<<(8192 * 16 * 32) / 256, 256, 0, stream>>>(kws, 1.0f);

    flash_attn_kernel<<<dim3(64, 32), 128, 0, stream>>>(qws, kws, vtws, ows);

    gemm_bf16_kernel<0><<<gGemm, 256, 0, stream>>>(ows, wob, bo, out, M, N, K);
}
